// LSTM_Model_89386859364651
// MI455X (gfx1250) — matrix operand order repaired and checked by reading, not compiled, not run on hardware
//
#include <hip/hip_runtime.h>

// ---------------------------------------------------------------------------
// Problem constants (from reference)
// ---------------------------------------------------------------------------
#define Tn 200
#define Bn 64
#define Dn 300
#define DP 320           // D padded to multiple of 32 for WMMA K
#define Hn 512
#define G4 2048          // 4*H
#define Ln 7

typedef __attribute__((ext_vector_type(16))) __bf16 bf16x16;
typedef __attribute__((ext_vector_type(8)))  float  f32x8;

// LDS row strides (halves), padded to break bank alignment on row stride
#define XSTR 328
#define HSTR 520

#define XBUF_BYTES (Bn * XSTR * 2)      // 41984
#define H_BYTES    (Bn * HSTR * 2)     // 66560
#define XBUF_OFF   0
#define H1_OFF     (XBUF_OFF + XBUF_BYTES)
#define H2_OFF     (H1_OFF + H_BYTES)
#define H3_OFF     (H2_OFF + H_BYTES)
#define LOG_OFF    (H3_OFF + H_BYTES)
#define SMEM_BYTES (LOG_OFF + Bn * 8 * 4)   // ~238 KB < 320 KB WGP LDS

union FragU { uint4 q[2]; bf16x16 v; };

// A-fragment (16x32 bf16) from LDS row-major [rows][stride halves].
// Contiguous k order (same as B tiles): lanes 0-15 row M=lane, halves k = 0..15;
// lanes 16-31 same M, halves k = 16..31.
__device__ __forceinline__ bf16x16 load_a_frag(const __bf16* base, int stride,
                                               int mrow, int kbase, int lane) {
  int k0 = kbase + ((lane & 16) ? 16 : 0);
  const __bf16* p = base + mrow * stride + k0;
  FragU f;
  f.q[0] = *(const uint4*)(p);        // halves k0..k0+7
  f.q[1] = *(const uint4*)(p + 8);    // halves k0+8..k0+15
  return f.v;
}

// B-fragment (32x16 bf16) from pre-tiled workspace: 32 lanes x 16 halves
// contiguous per tile (1024B). Lane holds column n=lane&15, K half by lane>=16.
__device__ __forceinline__ bf16x16 load_b_frag(const __bf16* tile, int lane) {
  const __bf16* p = tile + lane * 16;
  FragU f;
  f.q[0] = *(const uint4*)(p);
  f.q[1] = *(const uint4*)(p + 8);
  return f.v;
}

__device__ __forceinline__ float sigm(float x) {
  return __builtin_amdgcn_rcpf(1.0f + __expf(-x));
}
__device__ __forceinline__ float ssign(float x) {
  return x * __builtin_amdgcn_rcpf(1.0f + __builtin_fabsf(x));
}

// ---------------------------------------------------------------------------
// Weight retile: fp32 [K,N=2048] row-major  ->  bf16 WMMA-B tiles.
// dst dword layout: tile(nt*nkt+kt) * 256 + lane*8 + dword
//   n = nt*16 + (lane&15);  k = kt*32 + ((lane&16)?16:0) + 2*dword + {0,1}
// ---------------------------------------------------------------------------
__global__ void convert_weight(const float* __restrict__ src, __bf16* __restrict__ dst,
                               int Kreal, int nkt) {
  int idx = blockIdx.x * blockDim.x + threadIdx.x;     // dword index
  int total = 128 * nkt * 256;                         // 128 n-tiles (N=2048)
  if (idx >= total) return;
  int j2   = (idx & 7) * 2;
  int lane = (idx >> 3) & 31;
  int tile = idx >> 8;
  int kt = tile % nkt;
  int nt = tile / nkt;
  int n  = nt * 16 + (lane & 15);
  int k0 = kt * 32 + ((lane & 16) ? 16 : 0) + j2;
  float f0 = (k0     < Kreal) ? src[(size_t)k0       * G4 + n] : 0.0f;
  float f1 = (k0 + 1 < Kreal) ? src[(size_t)(k0 + 1) * G4 + n] : 0.0f;
  union { __bf16 h[2]; unsigned int u; } pk;
  pk.h[0] = (__bf16)f0;
  pk.h[1] = (__bf16)f1;
  ((unsigned int*)dst)[idx] = pk.u;
}

// ---------------------------------------------------------------------------
// Persistent fused LSTM: one workgroup, 32 waves. Wave w owns unit tile u=w
// (16 hidden units) across all 4 gates and all 4 batch tiles. h kept in LDS
// (bf16); c kept in workspace (f32, per-wave coalesced 1KB tiles, L2-resident)
// to stay under the 128-VGPR/wave budget at 8 waves/SIMD.
// ---------------------------------------------------------------------------
__global__ __launch_bounds__(1024, 1)
void lstm_persistent(const int* __restrict__ ids, const float* __restrict__ emb,
                     const __bf16* __restrict__ Wx1t, const __bf16* __restrict__ Wh1t,
                     const __bf16* __restrict__ Wx2t, const __bf16* __restrict__ Wh2t,
                     const __bf16* __restrict__ Wx3t, const __bf16* __restrict__ Wh3t,
                     const float* __restrict__ b1, const float* __restrict__ b2,
                     const float* __restrict__ b3,
                     float* __restrict__ cws,
                     const float* __restrict__ Wd0, const float* __restrict__ bd0,
                     const float* __restrict__ gamma, const float* __restrict__ beta,
                     const float* __restrict__ mmean, const float* __restrict__ mvar,
                     const float* __restrict__ alpha,
                     const float* __restrict__ Wd1, const float* __restrict__ bd1,
                     float* __restrict__ out) {
  __shared__ __align__(16) unsigned char smem[SMEM_BYTES];
  __bf16* xbuf = (__bf16*)(smem + XBUF_OFF);
  __bf16* hb0  = (__bf16*)(smem + H1_OFF);
  __bf16* hb1  = (__bf16*)(smem + H2_OFF);
  __bf16* hb2  = (__bf16*)(smem + H3_OFF);
  float*  logits = (float*)(smem + LOG_OFF);
  __bf16* hb[3] = { hb0, hb1, hb2 };

  const int tid  = threadIdx.x;
  const int lane = tid & 31;
  const int u    = tid >> 5;          // wave id == unit tile (0..31)
  const int ncol = lane & 15;
  const int hi8  = (lane & 16) ? 8 : 0;

  // zero h state in LDS
  for (int i = tid; i < (3 * H_BYTES) / 4; i += 1024)
    ((unsigned int*)(smem + H1_OFF))[i] = 0u;

  // zero this wave's cell-state tiles in workspace (each lane owns its slot)
  {
    f32x8 z;
#pragma unroll
    for (int j = 0; j < 8; ++j) z[j] = 0.0f;
#pragma unroll
    for (int l = 0; l < 3; ++l)
#pragma unroll
      for (int mb = 0; mb < 4; ++mb)
        *(f32x8*)(cws + (size_t)(((l * 4 + mb) * 32 + u) * 256) + lane * 8) = z;
  }

  const __bf16* wx[3] = { Wx1t, Wx2t, Wx3t };
  const __bf16* wh[3] = { Wh1t, Wh2t, Wh3t };
  const float*  bs[3] = { b1, b2, b3 };

  __syncthreads();

  for (int t = 0; t < Tn; ++t) {
    // ---- embedding gather -> xbuf (bf16, K padded to 320, row stride XSTR)
    for (int idx = tid; idx < Bn * (XSTR / 2); idx += 1024) {
      int b  = idx / (XSTR / 2);
      int c2 = idx % (XSTR / 2);
      int k  = c2 * 2;
      float f0 = 0.0f, f1 = 0.0f;
      if (k < Dn) {
        const float* row = emb + (size_t)ids[b * Tn + t] * Dn;
        f0 = row[k];
        if (k + 1 < Dn) f1 = row[k + 1];
      }
      union { __bf16 h[2]; unsigned int uu; } pk;
      pk.h[0] = (__bf16)f0;
      pk.h[1] = (__bf16)f1;
      ((unsigned int*)xbuf)[b * (XSTR / 2) + c2] = pk.uu;
    }
    __syncthreads();

#pragma unroll
    for (int l = 0; l < 3; ++l) {
      const __bf16* abase1 = (l == 0) ? xbuf : hb[l - 1];
      const int astr1 = (l == 0) ? XSTR : HSTR;
      const int nk1   = (l == 0) ? (DP / 32) : (Hn / 32);
      f32x8 hstage[4];

#pragma unroll
      for (int mb = 0; mb < 4; ++mb) {
        f32x8 acc[4];
#pragma unroll
        for (int gi = 0; gi < 4; ++gi) {
          float bv = bs[l][gi * Hn + u * 16 + ncol];
#pragma unroll
          for (int j = 0; j < 8; ++j) acc[gi][j] = bv;
        }
        const int mrow = mb * 16 + ncol;

        // input-part GEMM: x_t @ Wx (layer0) or h_{l-1} @ Wx
        for (int kt = 0; kt < nk1; ++kt) {
          bf16x16 a = load_a_frag(abase1, astr1, mrow, kt * 32, lane);
          bf16x16 bf[4];
#pragma unroll
          for (int gi = 0; gi < 4; ++gi)
            bf[gi] = load_b_frag(
                wx[l] + (size_t)((gi * 32 + u) * nk1 + kt) * 512, lane);
#pragma unroll
          for (int gi = 0; gi < 4; ++gi)
            acc[gi] = __builtin_amdgcn_wmma_f32_16x16x32_bf16(
                false, a, false, bf[gi], (short)0, acc[gi], false, false);
        }
        // recurrent-part GEMM: h_l @ Wh
        for (int kt = 0; kt < Hn / 32; ++kt) {
          bf16x16 a = load_a_frag(hb[l], HSTR, mrow, kt * 32, lane);
          bf16x16 bf[4];
#pragma unroll
          for (int gi = 0; gi < 4; ++gi)
            bf[gi] = load_b_frag(
                wh[l] + (size_t)((gi * 32 + u) * (Hn / 32) + kt) * 512, lane);
#pragma unroll
          for (int gi = 0; gi < 4; ++gi)
            acc[gi] = __builtin_amdgcn_wmma_f32_16x16x32_bf16(
                false, a, false, bf[gi], (short)0, acc[gi], false, false);
        }
        // gates: i,f,g,o (sigmoid / softsign per Keras cell); c from workspace
        float* ctile = cws + (size_t)(((l * 4 + mb) * 32 + u) * 256) + lane * 8;
        f32x8 cv = *(const f32x8*)ctile;
#pragma unroll
        for (int j = 0; j < 8; ++j) {
          float iv = sigm(acc[0][j]);
          float fv = sigm(acc[1][j]);
          float gv = ssign(acc[2][j]);
          float ov = sigm(acc[3][j]);
          float cn = fv * cv[j] + iv * gv;
          cv[j] = cn;
          hstage[mb][j] = ov * ssign(cn);
        }
        *(f32x8*)ctile = cv;
      }
      __syncthreads();   // all waves finished reading h_l of previous step
#pragma unroll
      for (int mb = 0; mb < 4; ++mb)
#pragma unroll
        for (int r = 0; r < 8; ++r)
          hb[l][(mb * 16 + r + hi8) * HSTR + u * 16 + ncol] =
              (__bf16)hstage[mb][r];
      __syncthreads();   // h_l(new) visible to layer l+1 / next step
    }
  }

  // ------------------------------------------------------------------
  // Head: d0 = h3 @ Wd0 + bd0 ; BN ; PReLU ; logits ; softmax
  // ------------------------------------------------------------------
  __syncthreads();
  __bf16* d0buf = xbuf;   // reuse xbuf LDS (32KB needed <= 41KB)
  for (int idx = tid; idx < Bn * 256; idx += 1024) {
    int b = idx >> 8;
    int j = idx & 255;
    float s = bd0[j];
    const __bf16* hrow = hb2 + b * HSTR;
    for (int k = 0; k < Hn; ++k) s += (float)hrow[k] * Wd0[k * 256 + j];
    float bn = (s - mmean[j]) * rsqrtf(mvar[j] + 1e-3f) * gamma[j] + beta[j];
    float pr = (bn >= 0.0f) ? bn : alpha[j] * bn;
    d0buf[idx] = (__bf16)pr;
  }
  __syncthreads();
  for (int idx = tid; idx < Bn * Ln; idx += 1024) {
    int b = idx / Ln;
    int l = idx % Ln;
    float s = bd1[l];
    for (int j = 0; j < 256; ++j) s += (float)d0buf[b * 256 + j] * Wd1[j * Ln + l];
    logits[b * 8 + l] = s;
  }
  __syncthreads();
  if (tid < Bn) {
    float m = -1e30f;
    for (int l = 0; l < Ln; ++l) m = fmaxf(m, logits[tid * 8 + l]);
    float e[Ln], ssum = 0.0f;
    for (int l = 0; l < Ln; ++l) { e[l] = __expf(logits[tid * 8 + l] - m); ssum += e[l]; }
    float r = __builtin_amdgcn_rcpf(ssum);
    for (int l = 0; l < Ln; ++l) out[tid * Ln + l] = e[l] * r;
  }
}

// ---------------------------------------------------------------------------
extern "C" void kernel_launch(void* const* d_in, const int* in_sizes, int n_in,
                              void* d_out, int out_size, void* d_ws, size_t ws_size,
                              hipStream_t stream) {
  (void)in_sizes; (void)n_in; (void)out_size; (void)ws_size;
  const int*   ids   = (const int*)d_in[0];
  const float* emb   = (const float*)d_in[1];
  const float* Wx1   = (const float*)d_in[2];
  const float* Wh1   = (const float*)d_in[3];
  const float* b1    = (const float*)d_in[4];
  const float* Wx2   = (const float*)d_in[5];
  const float* Wh2   = (const float*)d_in[6];
  const float* b2    = (const float*)d_in[7];
  const float* Wx3   = (const float*)d_in[8];
  const float* Wh3   = (const float*)d_in[9];
  const float* b3    = (const float*)d_in[10];
  const float* Wd0   = (const float*)d_in[11];
  const float* bd0   = (const float*)d_in[12];
  const float* gamma = (const float*)d_in[13];
  const float* beta  = (const float*)d_in[14];
  const float* mmean = (const float*)d_in[15];
  const float* mvar  = (const float*)d_in[16];
  const float* alpha = (const float*)d_in[17];
  const float* Wd1   = (const float*)d_in[18];
  const float* bd1   = (const float*)d_in[19];
  float* out = (float*)d_out;

  // Workspace: bf16 retiled weights (~11.3 MB) + f32 cell state (384 KB)
  __bf16* ws = (__bf16*)d_ws;
  const size_t szWx1 = (size_t)128 * 10 * 512;   // K=320 tiles
  const size_t szW   = (size_t)128 * 16 * 512;   // K=512 tiles
  __bf16* Wx1t = ws;
  __bf16* Wh1t = Wx1t + szWx1;
  __bf16* Wx2t = Wh1t + szW;
  __bf16* Wh2t = Wx2t + szW;
  __bf16* Wx3t = Wh2t + szW;
  __bf16* Wh3t = Wx3t + szW;
  float*  cws  = (float*)(Wh3t + szW);           // 3*4*32 tiles * 1KB

  struct { const float* s; __bf16* d; int kreal; int nkt; } cv[6] = {
    { Wx1, Wx1t, Dn, 10 }, { Wh1, Wh1t, Hn, 16 },
    { Wx2, Wx2t, Hn, 16 }, { Wh2, Wh2t, Hn, 16 },
    { Wx3, Wx3t, Hn, 16 }, { Wh3, Wh3t, Hn, 16 },
  };
  for (int i = 0; i < 6; ++i) {
    int total  = 128 * cv[i].nkt * 256;
    int blocks = (total + 255) / 256;
    convert_weight<<<dim3(blocks), dim3(256), 0, stream>>>(cv[i].s, cv[i].d,
                                                           cv[i].kreal, cv[i].nkt);
  }

  lstm_persistent<<<dim3(1), dim3(1024), 0, stream>>>(
      ids, emb, Wx1t, Wh1t, Wx2t, Wh2t, Wx3t, Wh3t, b1, b2, b3, cws,
      Wd0, bd0, gamma, beta, mmean, mvar, alpha, Wd1, bd1, out);
}
